// VectorQuantizer_12498354831611
// MI455X (gfx1250) — matrix operand order repaired and checked by reading, not compiled, not run on hardware
//
#include <hip/hip_runtime.h>
#include <hip/hip_bf16.h>

typedef __attribute__((ext_vector_type(16))) __bf16 v16bf;
typedef __attribute__((ext_vector_type(8)))  float  v8f;

#define D_DIM        256
#define K_CODES      1024
#define HW           1024        /* 32*32 */
#define TILE_CODES   16
#define NUM_TILES    (K_CODES / TILE_CODES)   /* 64 */
#define WAVES_PB     8
#define PIX_PER_WAVE 16
#define PIX_PER_BLK  128

struct alignas(16) u32x4 { unsigned int v[4]; };

__device__ __forceinline__ unsigned short f32_to_bf16(float f) {
    unsigned int u = __float_as_uint(f);
    u += 0x7FFFu + ((u >> 16) & 1u);        // round-to-nearest-even
    return (unsigned short)(u >> 16);
}

// ---------------------------------------------------------------------------
// Kernel 1: codebook -> bf16 copy + per-code squared norms
// ---------------------------------------------------------------------------
__global__ __launch_bounds__(256)
void vq_prep(const float* __restrict__ cb, unsigned short* __restrict__ cb16,
             float* __restrict__ cnorm) {
    const int k   = blockIdx.x;
    const int tid = threadIdx.x;
    float v = cb[(size_t)k * D_DIM + tid];
    cb16[(size_t)k * D_DIM + tid] = f32_to_bf16(v);
    float s = v * v;
    #pragma unroll
    for (int off = 16; off >= 1; off >>= 1) s += __shfl_xor(s, off, 32);
    __shared__ float ps[WAVES_PB];
    if ((tid & 31) == 0) ps[tid >> 5] = s;
    __syncthreads();
    if (tid == 0) {
        float t = 0.0f;
        #pragma unroll
        for (int i = 0; i < WAVES_PB; ++i) t += ps[i];
        cnorm[k] = t;
    }
}

// ---------------------------------------------------------------------------
// Kernel 2: distances via bf16 WMMA, argmin, gather, output, loss partials
// ---------------------------------------------------------------------------
__global__ __launch_bounds__(256)
void vq_main(const float* __restrict__ x, const float* __restrict__ cb,
             const unsigned short* __restrict__ cb16,
             const float* __restrict__ cnorm,
             float* __restrict__ out, float* __restrict__ partials) {
    __shared__ u32x4 s_tile[512];          // 16 codes x 256 dims bf16 = 8KB
    __shared__ float s_cnorm[K_CODES];     // 4KB
    __shared__ int   s_idx[PIX_PER_BLK];
    __shared__ float s_loss[WAVES_PB];

    const int tid  = threadIdx.x;
    const int lane = tid & 31;
    const int wv   = tid >> 5;
    const int hf   = lane >> 4;            // half-wave select
    const int ln16 = lane & 15;

    // stage all code norms into LDS once
    #pragma unroll
    for (int i = 0; i < 4; ++i) s_cnorm[tid + 256 * i] = cnorm[tid + 256 * i];

    const int n0    = blockIdx.x * PIX_PER_BLK + wv * PIX_PER_WAVE;
    const int batch = n0 >> 10;            // n = batch*1024 + hw
    const int hw0   = n0 & (HW - 1);
    const float* xb = x + (size_t)batch * (D_DIM * HW) + hw0 + ln16;

    // --- A fragments: 16 pixels x 256 dims, bf16, ISA 16-bit A 16x32 layout
    v16bf afrag[8];
    #pragma unroll
    for (int s = 0; s < 8; ++s) {
        union { unsigned short u16[16]; v16bf v; } ua;
        #pragma unroll
        for (int e = 0; e < 16; ++e) {
            int d = 32 * s + ((e >> 3) << 4) + hf * 8 + (e & 7);
            ua.u16[e] = f32_to_bf16(xb[(size_t)d * HW]);
        }
        afrag[s] = ua.v;
    }

    float minval[8];
    int   minidx[8];
    #pragma unroll
    for (int r = 0; r < 8; ++r) { minval[r] = 3.4e38f; minidx[r] = 0; }

    const u32x4* cbv4 = (const u32x4*)cb16;    // 32 u32x4 per code row
    for (int t = 0; t < NUM_TILES; ++t) {
        // cooperative stage of one 16-code tile (double-barrier reuse of LDS)
        u32x4 a0 = cbv4[(size_t)t * 512 + tid];
        u32x4 a1 = cbv4[(size_t)t * 512 + 256 + tid];
        __syncthreads();                       // previous tile reads done
        s_tile[tid]       = a0;
        s_tile[tid + 256] = a1;
        __syncthreads();                       // tile visible

        v8f acc = {0.f, 0.f, 0.f, 0.f, 0.f, 0.f, 0.f, 0.f};
        #pragma unroll
        for (int s = 0; s < 8; ++s) {
            union { u32x4 q[2]; v16bf v; } ub;
            const int base = ln16 * 32 + 4 * s + hf;   // B 32x16 layout
            ub.q[0] = s_tile[base];
            ub.q[1] = s_tile[base + 2];
            acc = __builtin_amdgcn_wmma_f32_16x16x32_bf16(
                false, afrag[s], false, ub.v, (short)0, acc, false, false);
        }

        const float cn   = s_cnorm[t * 16 + ln16];
        const int   code = t * 16 + ln16;
        #pragma unroll
        for (int r = 0; r < 8; ++r) {
            float sc = fmaf(-2.0f, acc[r], cn);     // d2 - ||x||^2 (const per row)
            bool better = (sc < minval[r]) || (sc == minval[r] && code < minidx[r]);
            minval[r] = better ? sc   : minval[r];
            minidx[r] = better ? code : minidx[r];
        }
    }

    // argmin butterfly within each 16-lane group (C layout: lane%16 = code col)
    #pragma unroll
    for (int off = 8; off >= 1; off >>= 1) {
        #pragma unroll
        for (int r = 0; r < 8; ++r) {
            float ov = __shfl_xor(minval[r], off, 32);
            int   oi = __shfl_xor(minidx[r], off, 32);
            bool better = (ov < minval[r]) || (ov == minval[r] && oi < minidx[r]);
            minval[r] = better ? ov : minval[r];
            minidx[r] = better ? oi : minidx[r];
        }
    }
    if (ln16 == 0) {
        #pragma unroll
        for (int r = 0; r < 8; ++r) s_idx[wv * 16 + r + 8 * hf] = minidx[r];
    }
    __syncthreads();

    // --- gather codebook rows, write NCHW output, accumulate loss
    const int p = ln16;                     // pixel within wave
    const int qh = hf;                      // which 128-dim half this lane does
    const int myidx = s_idx[wv * 16 + p];
    const float* cbr = cb  + (size_t)myidx * D_DIM;
    const float* xp  = x   + (size_t)batch * (D_DIM * HW) + hw0 + p;
    float*       op  = out + (size_t)batch * (D_DIM * HW) + hw0 + p;
    float lsum = 0.0f;
    #pragma unroll 2
    for (int j = 0; j < 32; ++j) {
        const int d = qh * 128 + j * 4;
        float4 c4 = *(const float4*)(cbr + d);
        float x0 = xp[(size_t)(d + 0) * HW];
        float x1 = xp[(size_t)(d + 1) * HW];
        float x2 = xp[(size_t)(d + 2) * HW];
        float x3 = xp[(size_t)(d + 3) * HW];
        op[(size_t)(d + 0) * HW] = c4.x;
        op[(size_t)(d + 1) * HW] = c4.y;
        op[(size_t)(d + 2) * HW] = c4.z;
        op[(size_t)(d + 3) * HW] = c4.w;
        float e0 = c4.x - x0, e1 = c4.y - x1, e2 = c4.z - x2, e3 = c4.w - x3;
        lsum += e0 * e0 + e1 * e1 + e2 * e2 + e3 * e3;
    }
    #pragma unroll
    for (int off = 16; off >= 1; off >>= 1) lsum += __shfl_xor(lsum, off, 32);
    if (lane == 0) s_loss[wv] = lsum;
    __syncthreads();
    if (tid == 0) {
        float tot = 0.0f;
        #pragma unroll
        for (int i = 0; i < WAVES_PB; ++i) tot += s_loss[i];
        partials[blockIdx.x] = tot;
    }
}

// ---------------------------------------------------------------------------
// Kernel 3: deterministic reduction of per-block partials -> commitment loss
// ---------------------------------------------------------------------------
__global__ __launch_bounds__(256)
void vq_loss_finalize(const float* __restrict__ partials, int nparts,
                      float scale, float* __restrict__ dst) {
    __shared__ float s[WAVES_PB];
    const int tid = threadIdx.x;
    float v = 0.0f;
    for (int i = tid; i < nparts; i += 256) v += partials[i];
    #pragma unroll
    for (int off = 16; off >= 1; off >>= 1) v += __shfl_xor(v, off, 32);
    if ((tid & 31) == 0) s[tid >> 5] = v;
    __syncthreads();
    if (tid == 0) {
        float tot = 0.0f;
        #pragma unroll
        for (int i = 0; i < WAVES_PB; ++i) tot += s[i];
        dst[0] = tot * scale;
    }
}

// ---------------------------------------------------------------------------
extern "C" void kernel_launch(void* const* d_in, const int* in_sizes, int n_in,
                              void* d_out, int out_size, void* d_ws, size_t ws_size,
                              hipStream_t stream) {
    const float* x  = (const float*)d_in[0];   // (64,256,32,32) fp32 NCHW
    const float* cb = (const float*)d_in[1];   // (1024,256) fp32
    float* out = (float*)d_out;                // 64*256*32*32 q_st + 1 loss

    const int n_elem  = in_sizes[0];           // 16777216
    const int npix    = n_elem / D_DIM;        // 65536
    const int nblocks = npix / PIX_PER_BLK;    // 512

    unsigned short* cb16  = (unsigned short*)d_ws;                       // 512KB
    float* cnorm    = (float*)((char*)d_ws + (size_t)K_CODES * D_DIM * 2);
    float* partials = (float*)((char*)d_ws + (size_t)K_CODES * D_DIM * 2
                               + (size_t)K_CODES * 4);

    vq_prep<<<K_CODES, 256, 0, stream>>>(cb, cb16, cnorm);
    vq_main<<<nblocks, 256, 0, stream>>>(x, cb, cb16, cnorm, out, partials);
    const float scale = 1.25f / (float)n_elem;
    vq_loss_finalize<<<1, 256, 0, stream>>>(partials, nblocks, scale,
                                            out + (size_t)n_elem);
}
